// VeduBitNetCNN_25786983645445
// MI455X (gfx1250) — matrix operand order repaired and checked by reading, not compiled, not run on hardware
//
#include <hip/hip_runtime.h>
#include <hip/hip_bf16.h>

// ---------------------------------------------------------------------------
// BitNet CNN forward for MI455X (gfx1250, wave32, WMMA bf16 16x16x32).
// Bandwidth-bound (~120MB min traffic @23.3TB/s): bf16 intermediates, fused
// bias/relu/pool, WMMA for conv2/fc1/fc2 with b128 fragment loads, and
// double-buffered ASYNC global->LDS staging (ASYNCcnt) in the fc1 GEMM.
//
// Fragment layouts (ISA 7.12.2, 16-bit):
//   A 16x32: lane holds M=lane&15; halves = two contiguous K-runs
//            [lhi*8 .. +7] and [16+lhi*8 .. +7]  (lhi = lane>>4)
//   B 32x16: lane holds N=lane&15; halves = two K-runs [lhi*8 .. +7] and [16+lhi*8 .. +7]
// => A frag = uint4 pair {rp[lhi], rp[2+lhi]} from a [M][32] row-major tile
// => B frag = uint4 pair {bp[0], bp[2]} at K = lhi*8 of an [N][K] row-major tile
// ---------------------------------------------------------------------------

typedef __attribute__((ext_vector_type(16))) __bf16 v16bf;
typedef __attribute__((ext_vector_type(8)))  float  v8f;

#define EPSQ 1e-8f

union FragU { uint4 q[2]; v16bf v; };

__device__ __forceinline__ v8f wmma_bf16(v16bf a, v16bf b, v8f c) {
  return __builtin_amdgcn_wmma_f32_16x16x32_bf16(false, a, false, b, (short)0, c,
                                                 false, false);
}
__device__ __forceinline__ v8f v8f_zero() {
  v8f z = {0.f, 0.f, 0.f, 0.f, 0.f, 0.f, 0.f, 0.f};
  return z;
}
__device__ __forceinline__ uint4 uint4_zero() {
  uint4 z; z.x = z.y = z.z = z.w = 0u; return z;
}

// fragkA still needed for fc2's compute-on-load A fragment
__device__ __forceinline__ int fragkA(int lane, int e) {
  int v = e >> 1;
  return (((v & 3) << 1) | (e & 1)) + (((v >> 2) & 1) << 4) + (((lane >> 4) & 1) << 3);
}

// Async global->LDS b128 copy (CDNA5 TDM-adjacent async path, ASYNCcnt).
// VFLAT encoding: VDST = LDS byte address VGPR, VADDR = 64-bit global address.
__device__ __forceinline__ void async_ld_b128(void* lds, const void* g) {
  unsigned l = (unsigned)(size_t)lds;  // flat LDS addr low bits = LDS offset
  unsigned long long ga = (unsigned long long)(size_t)g;
  asm volatile("global_load_async_to_lds_b128 %0, %1, off"
               :: "v"(l), "v"(ga) : "memory");
}

// ---------------------------------------------------------------------------
// 0) init: zero scale accumulators, fc1 split-K output, padded fc2 weights
// ---------------------------------------------------------------------------
__global__ void k_init(float* acc, float* fc1out, unsigned* wqf2u) {
  int i = blockIdx.x * 256 + threadIdx.x;
  if (i < 4) acc[i] = 0.f;
  if (i < 512 * 128) fc1out[i] = 0.f;
  if (i < 1024) wqf2u[i] = 0u;  // 2048 bf16 of padded fc2 weight [16][128]
}

// ---------------------------------------------------------------------------
// 1) absmean partial sums: sum(|w|) -> atomicAdd(acc)
// ---------------------------------------------------------------------------
__global__ void k_absum(const float* __restrict__ w, int n, float* acc) {
  __shared__ float red[256];
  float s = 0.f;
  for (int i = blockIdx.x * 256 + threadIdx.x; i < n; i += gridDim.x * 256)
    s += fabsf(w[i]);
  red[threadIdx.x] = s;
  __syncthreads();
  for (int off = 128; off > 0; off >>= 1) {
    if (threadIdx.x < off) red[threadIdx.x] += red[threadIdx.x + off];
    __syncthreads();
  }
  if (threadIdx.x == 0) atomicAdd(acc, red[0]);
}

// ---------------------------------------------------------------------------
// 2) quantize: wq = clip(round(w/(s+eps)),-1,1)*s, s = mean(|w|), to bf16
// ---------------------------------------------------------------------------
__device__ __forceinline__ __bf16 qtern(float w, float s) {
  float t = roundf(w / (s + EPSQ));
  t = fminf(1.f, fmaxf(-1.f, t));
  return (__bf16)(t * s);
}

__global__ void k_quant(const float* __restrict__ w, __bf16* dst, int n,
                        const float* acc, float inv_n) {
  int i = blockIdx.x * 256 + threadIdx.x;
  if (i >= n) return;
  float s = acc[0] * inv_n;
  dst[i] = qtern(w[i], s);
}

// ---------------------------------------------------------------------------
// 3) conv1 (3->16, 3x3, pad 1) + bias + relu + 2x2 maxpool, direct fp32 FMA.
//    grid: (6,6,512) tiles of 8x8 pooled outputs.
//    out NHWC bf16 [512][48][48][16], packed 4-channel b64 stores.
// ---------------------------------------------------------------------------
__global__ __launch_bounds__(256) void k_conv1(const float* __restrict__ x,
                                               const __bf16* __restrict__ wq1,
                                               const float* __restrict__ b1,
                                               __bf16* __restrict__ act1) {
  __shared__ float xs[3][18][18];   // 16x16 input tile + halo
  __shared__ float wsh[16][27];
  __shared__ float bsh[16];
  const int n = blockIdx.z, ty = blockIdx.y, tx = blockIdx.x;
  const int tid = threadIdx.x;
  for (int i = tid; i < 3 * 18 * 18; i += 256) {
    int ci = i / 324, rem = i - ci * 324, r = rem / 18, c = rem - r * 18;
    int gy = ty * 16 - 1 + r, gx = tx * 16 - 1 + c;
    float v = 0.f;
    if (gy >= 0 && gy < 96 && gx >= 0 && gx < 96)
      v = x[((n * 3 + ci) * 96 + gy) * 96 + gx];
    xs[ci][r][c] = v;
  }
  for (int i = tid; i < 16 * 27; i += 256) wsh[i / 27][i % 27] = (float)wq1[i];
  if (tid < 16) bsh[tid] = b1[tid];
  __syncthreads();
  // 64 positions x 4 channel-groups: thread -> (pos, cg), 4 outputs each
  const int pos = tid >> 2, cg = tid & 3;
  const int py = pos >> 3, px = pos & 7;
  union { __bf16 h[4]; uint2 u; } pk;
#pragma unroll
  for (int c = 0; c < 4; ++c) {
    const int co = cg * 4 + c;
    float m = -1e30f;
#pragma unroll
    for (int dy = 0; dy < 2; ++dy)
#pragma unroll
      for (int dx = 0; dx < 2; ++dx) {
        int yy = py * 2 + dy, xx = px * 2 + dx;
        float a = bsh[co];
#pragma unroll
        for (int ci = 0; ci < 3; ++ci)
#pragma unroll
          for (int ky = 0; ky < 3; ++ky)
#pragma unroll
            for (int kx = 0; kx < 3; ++kx)
              a += xs[ci][yy + ky][xx + kx] * wsh[co][ci * 9 + ky * 3 + kx];
        m = fmaxf(m, a);
      }
    pk.h[c] = (__bf16)fmaxf(m, 0.f);
  }
  const int gpy = ty * 8 + py, gpx = tx * 8 + px;
  *(uint2*)(act1 + (((size_t)n * 48 + gpy) * 48 + gpx) * 16 + cg * 4) = pk.u;
}

// ---------------------------------------------------------------------------
// 4) conv2 (16->32, 3x3, pad 1) implicit-GEMM WMMA + bias + relu + pool.
//    One WG per (image, output row-pair): M = 96 positions, N = 32 channels,
//    K = 144 (9 taps x 16 ci, tap-major) padded to 160.
//    NHWC input tile in LDS => every A-fragment K-run is 16 contiguous bytes.
//    4 waves; wave w: Ntile = w&1, Mtiles = 3*(w>>1)..+2.
//    Pool fused via LDS; out NCHW bf16 [512][32][24][24] (matches flatten).
// ---------------------------------------------------------------------------
__global__ __launch_bounds__(128) void k_conv2(const __bf16* __restrict__ act1,
                                               const __bf16* __restrict__ wq2,
                                               const float* __restrict__ b2,
                                               __bf16* __restrict__ act2) {
  __shared__ __attribute__((aligned(16))) __bf16 at[4][50][16]; // rows y0-1..y0+2, cols -1..48, NHWC
  __shared__ __attribute__((aligned(16))) __bf16 wl[32][160];   // [o][k=tap*16+ci], zero padded
  __shared__ float pool[96][32];
  __shared__ float bsh[32];
  const int rp = blockIdx.x, n = blockIdx.y;
  const int tid = threadIdx.x;
  const int y0 = rp * 2;
  // NHWC tile fill: 200 pixels x 2 uint4 (16 ci = 32B)
  for (int i = tid; i < 400; i += 128) {
    int p = i >> 1, h = i & 1;
    int lr = p / 50, lc = p - lr * 50;
    int gy = y0 - 1 + lr, gx = lc - 1;
    uint4 v = uint4_zero();
    if (gy >= 0 && gy < 48 && gx >= 0 && gx < 48)
      v = *(const uint4*)(act1 + (((size_t)n * 48 + gy) * 48 + gx) * 16 + h * 8);
    *(uint4*)&at[lr][lc][h * 8] = v;
  }
  // weights: wl[o][tap*16+ci] = wq2[o][ci][tap], zeros for k>=144
  for (int i = tid; i < 32 * 160; i += 128) {
    int o = i / 160, k = i - o * 160;
    int t = k >> 4, ci = k & 15;
    wl[o][k] = (t < 9) ? wq2[o * 144 + ci * 9 + t] : (__bf16)0.f;
  }
  if (tid < 32) bsh[tid] = b2[tid];
  __syncthreads();

  const int wv = tid >> 5, lane = tid & 31;
  const int ln = lane & 15, lhi = (lane >> 4) & 1;
  const int nb = (wv & 1) * 16;
  const int mb = (wv >> 1) * 3;
  v8f acc[3];
#pragma unroll
  for (int j = 0; j < 3; ++j) acc[j] = v8f_zero();

#pragma unroll
  for (int kc = 0; kc < 5; ++kc) {
    // B fragment: K-runs [kc*32 + lhi*8 ..+7] and [kc*32 + 16 + lhi*8 ..+7] of wl[nb+ln]
    FragU bfu;
    {
      const uint4* bp = (const uint4*)&wl[nb + ln][kc * 32 + lhi * 8];
      bfu.q[0] = bp[0];
      bfu.q[1] = bp[2];
    }
#pragma unroll
    for (int j = 0; j < 3; ++j) {
      int p = (mb + j) * 16 + ln;  // position 0..95
      int row = (p >= 48) ? 1 : 0;
      int col = p - row * 48;
      FragU afu;
      {
        int K0 = kc * 32 + lhi * 8;          // < 144 always
        int t0 = K0 >> 4, c0 = K0 & 15;
        int ky0 = t0 / 3, kx0 = t0 - ky0 * 3;
        afu.q[0] = *(const uint4*)&at[row + ky0][col + kx0][c0];
        int K1 = K0 + 16;                    // >= 144 only when kc==4
        if (K1 < 144) {
          int t1 = K1 >> 4, c1 = K1 & 15;
          int ky1 = t1 / 3, kx1 = t1 - ky1 * 3;
          afu.q[1] = *(const uint4*)&at[row + ky1][col + kx1][c1];
        } else {
          afu.q[1] = uint4_zero();
        }
      }
      acc[j] = wmma_bf16(afu.v, bfu.v, acc[j]);
    }
  }
  // D layout: M = r + lhi*8, N = ln -> pool buffer
#pragma unroll
  for (int j = 0; j < 3; ++j)
#pragma unroll
    for (int r = 0; r < 8; ++r)
      pool[(mb + j) * 16 + r + lhi * 8][nb + ln] = acc[j][r];
  __syncthreads();
  for (int i = tid; i < 24 * 32; i += 128) {
    int px = i >> 5, c = i & 31;
    float m = fmaxf(fmaxf(pool[2 * px][c], pool[2 * px + 1][c]),
                    fmaxf(pool[48 + 2 * px][c], pool[49 + 2 * px][c]));
    float v = fmaxf(m + bsh[c], 0.f);
    act2[((n * 32 + c) * 24 + rp) * 24 + px] = (__bf16)v;
  }
}

// ---------------------------------------------------------------------------
// 5) fc1: [512,18432]bf16 x wq[128,18432]bf16([N][K]) -> fp32, split-K.
//    grid (64 ksplits, 4 mstripes), 8 waves; wave w owns Mtile w, 8 Ntiles.
//    K per WG = 288 (9 chunks of 32). Double-buffered ASYNC global->LDS tile
//    staging (4 async b128 instrs/wave/chunk, in-order completion =>
//    s_wait_asynccnt 4 retires exactly the current chunk while the next one
//    stays in flight). Fragments = b128 pairs. Epilogue: global atomicAdd f32.
// ---------------------------------------------------------------------------
__global__ __launch_bounds__(256) void k_fc1(const __bf16* __restrict__ act2,
                                             const __bf16* __restrict__ wq,
                                             float* __restrict__ fc1out) {
  __shared__ __attribute__((aligned(16))) __bf16 at[2][128][32];  // [buf][M][K]
  __shared__ __attribute__((aligned(16))) __bf16 bt[2][128][32];  // [buf][N][K]
  const int mbase = blockIdx.y * 128;
  const int kbase = blockIdx.x * 288;
  const int tid = threadIdx.x, wv = tid >> 5, lane = tid & 31;
  const int ln = lane & 15, lhi = (lane >> 4) & 1;

  // stage one K-chunk into buffer b: 512 b128 per tile, 2 per thread per tile
  auto issue = [&](int kc, int b) {
#pragma unroll
    for (int i = tid; i < 512; i += 256) {
      int m = i >> 2, q = i & 3;
      async_ld_b128(&((uint4*)&at[b][m][0])[q],
                    act2 + (size_t)(mbase + m) * 18432 + kbase + kc * 32 + q * 8);
      async_ld_b128(&((uint4*)&bt[b][m][0])[q],
                    wq + (size_t)m * 18432 + kbase + kc * 32 + q * 8);
    }
  };

  v8f acc[8];
#pragma unroll
  for (int j = 0; j < 8; ++j) acc[j] = v8f_zero();

  issue(0, 0);
  issue(1, 1);
  for (int kc = 0; kc < 9; ++kc) {
    if (kc < 8)
      asm volatile("s_wait_asynccnt 0x4" ::: "memory");  // chunk kc arrived
    else
      asm volatile("s_wait_asynccnt 0x0" ::: "memory");
    __syncthreads();
    const int b = kc & 1;
    FragU afu;
    {
      const uint4* rp = (const uint4*)&at[b][wv * 16 + ln][0];
      afu.q[0] = rp[lhi];
      afu.q[1] = rp[2 + lhi];
    }
#pragma unroll
    for (int j = 0; j < 8; ++j) {
      FragU bfu;
      const uint4* bp = (const uint4*)&bt[b][j * 16 + ln][lhi * 8];
      bfu.q[0] = bp[0];
      bfu.q[1] = bp[2];
      acc[j] = wmma_bf16(afu.v, bfu.v, acc[j]);
    }
    __syncthreads();  // all waves done reading buf b before it is re-filled
    if (kc + 2 <= 8) issue(kc + 2, b);
  }
#pragma unroll
  for (int j = 0; j < 8; ++j)
#pragma unroll
    for (int r = 0; r < 8; ++r)
      atomicAdd(&fc1out[(size_t)(mbase + wv * 16 + r + lhi * 8) * 128 + j * 16 + ln],
                acc[j][r]);
}

// ---------------------------------------------------------------------------
// 6) fc2: fuses fc1 bias+relu on A load; [512,128] x wq[16,128]([N][K]) WMMA.
//    grid: 32 Mtiles, 1 wave each. Writes d_out [512][10] fp32.
// ---------------------------------------------------------------------------
__global__ __launch_bounds__(32) void k_fc2(const float* __restrict__ fc1out,
                                            const float* __restrict__ fb1,
                                            const __bf16* __restrict__ w2q,
                                            const float* __restrict__ fb2,
                                            float* __restrict__ out) {
  const int mt = blockIdx.x;
  const int lane = threadIdx.x, ln = lane & 15, lhi = (lane >> 4) & 1;
  v8f acc = v8f_zero();
#pragma unroll
  for (int kc = 0; kc < 4; ++kc) {
    v16bf afr;
#pragma unroll
    for (int e = 0; e < 16; ++e) {
      int ka = kc * 32 + fragkA(lane, e);
      float v = fc1out[(mt * 16 + ln) * 128 + ka] + fb1[ka];
      afr[e] = (__bf16)fmaxf(v, 0.f);
    }
    FragU bfu;
    const uint4* bp = (const uint4*)(w2q + ln * 128 + kc * 32 + lhi * 8);
    bfu.q[0] = bp[0];
    bfu.q[1] = bp[2];
    acc = wmma_bf16(afr, bfu.v, acc);
  }
  if (ln < 10) {
#pragma unroll
    for (int r = 0; r < 8; ++r)
      out[(mt * 16 + r + lhi * 8) * 10 + ln] = acc[r] + fb2[ln];
  }
}

// ---------------------------------------------------------------------------
// host launcher
// ---------------------------------------------------------------------------
extern "C" void kernel_launch(void* const* d_in, const int* in_sizes, int n_in,
                              void* d_out, int out_size, void* d_ws, size_t ws_size,
                              hipStream_t stream) {
  const float* x   = (const float*)d_in[0];   // [512,3,96,96]
  const float* w1  = (const float*)d_in[1];   // [16,3,3,3]
  const float* b1  = (const float*)d_in[2];   // [16]
  const float* w2  = (const float*)d_in[3];   // [32,16,3,3]
  const float* b2  = (const float*)d_in[4];   // [32]
  const float* wf1 = (const float*)d_in[5];   // [128,18432]
  const float* bf1 = (const float*)d_in[6];   // [128]
  const float* wf2 = (const float*)d_in[7];   // [10,128]
  const float* bf2 = (const float*)d_in[8];   // [10]
  (void)in_sizes; (void)n_in; (void)out_size; (void)ws_size;

  char* ws = (char*)d_ws;
  size_t off = 0;
  auto wsalloc = [&](size_t bytes) -> void* {
    void* p = ws + off;
    off = (off + bytes + 255) & ~(size_t)255;
    return p;
  };
  float*  acc    = (float*)wsalloc(4 * sizeof(float));
  __bf16* wq1    = (__bf16*)wsalloc((size_t)432 * 2);
  __bf16* wq2    = (__bf16*)wsalloc((size_t)4608 * 2);
  __bf16* wqf2   = (__bf16*)wsalloc((size_t)2048 * 2);        // [16][128] padded
  __bf16* wqf1   = (__bf16*)wsalloc((size_t)128 * 18432 * 2); // [N][K]
  float*  fc1out = (float*)wsalloc((size_t)512 * 128 * 4);
  __bf16* act1   = (__bf16*)wsalloc((size_t)512 * 48 * 48 * 16 * 2); // NHWC
  __bf16* act2   = (__bf16*)wsalloc((size_t)512 * 32 * 24 * 24 * 2); // NCHW
  // total ~62 MB

  k_init<<<256, 256, 0, stream>>>(acc, fc1out, (unsigned*)wqf2);

  k_absum<<<2, 256, 0, stream>>>(w1, 432, acc + 0);
  k_absum<<<18, 256, 0, stream>>>(w2, 4608, acc + 1);
  k_absum<<<4096, 256, 0, stream>>>(wf1, 128 * 18432, acc + 2);
  k_absum<<<5, 256, 0, stream>>>(wf2, 1280, acc + 3);

  k_quant<<<2, 256, 0, stream>>>(w1, wq1, 432, acc + 0, 1.f / 432.f);
  k_quant<<<18, 256, 0, stream>>>(w2, wq2, 4608, acc + 1, 1.f / 4608.f);
  k_quant<<<(128 * 18432 + 255) / 256, 256, 0, stream>>>(
      wf1, wqf1, 128 * 18432, acc + 2, 1.f / (128.f * 18432.f));
  k_quant<<<5, 256, 0, stream>>>(wf2, wqf2, 1280, acc + 3, 1.f / 1280.f);

  k_conv1<<<dim3(6, 6, 512), 256, 0, stream>>>(x, wq1, b1, act1);
  k_conv2<<<dim3(24, 512), 128, 0, stream>>>(act1, wq2, b2, act2);
  k_fc1<<<dim3(64, 4), 256, 0, stream>>>(act2, wqf1, fc1out);
  k_fc2<<<32, 32, 0, stream>>>(fc1out, bf1, wqf2, bf2, (float*)d_out);
}
